// UV_Aggregator_90829968376430
// MI455X (gfx1250) — matrix operand order repaired and checked by reading, not compiled, not run on hardware
//
#include <hip/hip_runtime.h>
#include <hip/hip_bf16.h>
#include <math.h>

// ---------------------------------------------------------------------------
// GraphRec UV-Aggregator for MI455X (gfx1250, wave32, WMMA bf16 16x16x32)
// One workgroup (256 thr = 8 waves) per batch row. All GEMMs run from LDS.
// ---------------------------------------------------------------------------

typedef __bf16 bf16_t;
typedef bf16_t v16bf __attribute__((ext_vector_type(16)));
typedef bf16_t v8bf  __attribute__((ext_vector_type(8)));
typedef bf16_t v4bf  __attribute__((ext_vector_type(4)));
typedef float  v8f   __attribute__((ext_vector_type(8)));

#define L_HIST  50
#define DM      64
#define LDX     136   // stride for K=128 bf16 buffers (272B rows, 16B aligned, conflict-free)
#define LDH     72    // stride for K=64  bf16 buffers (144B rows)

// f32x4 -> bf16x4, one ds_store_b64
__device__ inline void cvt_store_bf16x4(bf16_t* dst, float4 v) {
    v4bf o;
    o[0] = (bf16_t)v.x; o[1] = (bf16_t)v.y; o[2] = (bf16_t)v.z; o[3] = (bf16_t)v.w;
    *(v4bf*)dst = o;
}

// A-fragment (16x32 bf16, M x K). ISA 7.12.2: lanes 0-15 -> M=lane, K={0..7,16..23};
// lanes 16-31 -> M=lane-16, K={8..15,24..31}. Two 16-byte contiguous LDS loads.
__device__ inline v16bf lds_load_a(const bf16_t* tile, int lda, int lane) {
    const int half = lane >> 4;
    const int m    = lane & 15;
    const bf16_t* row = tile + m * lda;
    union { v16bf v; v8bf h[2]; } u;
    u.h[0] = *(const v8bf*)(row + 8 * half);
    u.h[1] = *(const v8bf*)(row + 16 + 8 * half);
    return u.v;
}

// B-fragment from row-major (N_out x K) weights: lane n = output channel,
// low lanes hold K={0..7,16..23}, high lanes K={8..15,24..31} (matches A).
__device__ inline v16bf lds_load_b(const bf16_t* tile, int ldb, int lane) {
    const int half = lane >> 4;
    const int n    = lane & 15;
    const bf16_t* row = tile + n * ldb;
    union { v16bf v; v8bf h[2]; } u;
    u.h[0] = *(const v8bf*)(row + 8 * half);
    u.h[1] = *(const v8bf*)(row + 16 + 8 * half);
    return u.v;
}

// One fused GEMM phase: D = relu(A(64xK) * W(64xK)^T + bias), 4x4 tiles of 16x16,
// 8 waves -> 2 tiles each (same nt column -> B fragments stay in VGPRs).
// All fragment loads are issued before the WMMA chain (single dscnt wait).
template <int KSTEPS>
__device__ inline void gemm_relu(const bf16_t* __restrict__ A, int lda,
                                 const bf16_t* __restrict__ W, int ldw,
                                 const float*  __restrict__ bias,
                                 bf16_t* Dbf, int lddb,
                                 float*  Df,  int lddf,
                                 int wid, int lane) {
    const int half = lane >> 4;
    const int col  = lane & 15;
    for (int t = wid; t < 16; t += 8) {
        const int mt = t >> 2;
        const int nt = t & 3;
        v16bf av[KSTEPS], bv[KSTEPS];
#pragma unroll
        for (int k = 0; k < KSTEPS; ++k) {
            av[k] = lds_load_a(A + mt * 16 * lda + k * 32, lda, lane);
            bv[k] = lds_load_b(W + nt * 16 * ldw + k * 32, ldw, lane);
        }
        v8f acc = {};
#pragma unroll
        for (int k = 0; k < KSTEPS; ++k)
            acc = __builtin_amdgcn_wmma_f32_16x16x32_bf16(
                false, av[k], false, bv[k], (short)0, acc, false, false);

        const int   gn = nt * 16 + col;
        const float bb = bias[gn];
#pragma unroll
        for (int r = 0; r < 8; ++r) {
            float v = acc[r] + bb;
            v = v > 0.f ? v : 0.f;
            const int gm = mt * 16 + r + 8 * half;   // C/D layout: VGPR r -> M=r (+8 hi half)
            if (Dbf) Dbf[gm * lddb + gn] = (bf16_t)v;
            if (Df)  Df[gm * lddf + gn]  = v;
        }
    }
}

__global__ __launch_bounds__(256)
void uv_aggregator_kernel(const int* __restrict__ nodes,
                          const int* __restrict__ h_uv,
                          const int* __restrict__ h_r,
                          const float* __restrict__ v2e,
                          const float* __restrict__ u2e,
                          const float* __restrict__ r2e,
                          const float* __restrict__ w1,  const float* __restrict__ b1,
                          const float* __restrict__ w2,  const float* __restrict__ b2,
                          const float* __restrict__ a1w, const float* __restrict__ a1b,
                          const float* __restrict__ a2w, const float* __restrict__ a2b,
                          const float* __restrict__ a3w,
                          float* __restrict__ out) {
    const int b    = blockIdx.x;
    const int tid  = threadIdx.x;
    const int lane = tid & 31;
    const int wid  = tid >> 5;

    // LDS (~72KB): buffers are reused across phases.
    __shared__ alignas(16) bf16_t sX [64 * LDX];  // phase1 A = [e_uv;e_r]; later XA = [O;uv_rep]
    __shared__ alignas(16) bf16_t sW1[64 * LDX];  // W1, later att1 weights
    __shared__ alignas(16) bf16_t sH [64 * LDH];  // H, later att1 output
    __shared__ alignas(16) bf16_t sW2[64 * LDH];  // W2, later att2 weights
    __shared__ alignas(16) float  sOf[64 * DM];   // O in f32 for final reduction
    __shared__ float  sBias[4 * DM];              // b1,b2,a1b,a2b
    __shared__ float  sA3[DM];
    __shared__ alignas(16) bf16_t sUV[DM];
    __shared__ int    sIu[64];
    __shared__ int    sIr[64];
    __shared__ float  sS[64];
    __shared__ float  sWt[64];

    // ---------------- Phase -1: stage indices once --------------------------
    if (tid < 64) {
        sIu[tid] = (tid < L_HIST) ? h_uv[b * L_HIST + tid] : 0;
        sIr[tid] = (tid < L_HIST) ? h_r [b * L_HIST + tid] : 0;
        sS[tid]  = 0.f;
    }
    __syncthreads();

    // ---------------- Phase 0: gathers + weight staging (f32x4 -> bf16x4) ---
    if (tid < 64) {
        sBias[tid]       = b1[tid];
        sBias[64 + tid]  = b2[tid];
        sBias[128 + tid] = a1b[tid];
        sBias[192 + tid] = a2b[tid];
        sA3[tid]         = a3w[tid];
        sUV[tid]         = (bf16_t)u2e[(size_t)nodes[b] * DM + tid];
    }
    for (int j = tid; j < 64 * 32; j += 256) {       // X = [e_uv ; e_r], rows 50..63 = 0
        const int row = j >> 5, c4 = (j & 31) << 2;
        float4 v = make_float4(0.f, 0.f, 0.f, 0.f);
        if (row < L_HIST) {
            if (c4 < DM) v = *(const float4*)(v2e + (size_t)sIu[row] * DM + c4);
            else         v = *(const float4*)(r2e + (size_t)sIr[row] * DM + (c4 - DM));
        }
        cvt_store_bf16x4(sX + row * LDX + c4, v);
    }
    for (int j = tid; j < 2048; j += 256) {          // W1 (64x128) = 2048 float4
        const int row = j >> 5, c4 = (j & 31) << 2;
        cvt_store_bf16x4(sW1 + row * LDX + c4, ((const float4*)w1)[j]);
    }
    for (int j = tid; j < 1024; j += 256) {          // W2 (64x64) = 1024 float4
        const int row = j >> 4, c4 = (j & 15) << 2;
        cvt_store_bf16x4(sW2 + row * LDH + c4, ((const float4*)w2)[j]);
    }
    __syncthreads();

    // ---------------- Phase 1: H = relu(X @ W1^T + b1) ----------------------
    gemm_relu<4>(sX, LDX, sW1, LDX, sBias + 0, sH, LDH, (float*)nullptr, 0, wid, lane);
    __syncthreads();

    // ---------------- Phase 2: O = relu(H @ W2^T + b2) ----------------------
    // O(bf16) -> XA[:,0:64]; O(f32) -> sOf. Overlapped: fill XA[:,64:128] with
    // uv_rep and stream att1 weights into the now-dead sW1 buffer.
    gemm_relu<2>(sH, LDH, sW2, LDH, sBias + 64, sX /*XA cols 0..63*/, LDX, sOf, DM, wid, lane);
    for (int j = tid; j < 64 * 16; j += 256) {       // XA[:,64:128] = uv_rep (bf16x4 copies)
        const int row = j >> 4, c4 = (j & 15) << 2;
        *(v4bf*)(sX + row * LDX + DM + c4) = *(const v4bf*)(sUV + c4);
    }
    for (int j = tid; j < 2048; j += 256) {          // att1 weights (64x128)
        const int row = j >> 5, c4 = (j & 31) << 2;
        cvt_store_bf16x4(sW1 + row * LDX + c4, ((const float4*)a1w)[j]);
    }
    __syncthreads();

    // ---------------- Phase 3: A = relu(XA @ att1^T + a1b) ------------------
    // Overlapped: stream att2 weights into the now-dead sW2 buffer.
    gemm_relu<4>(sX, LDX, sW1, LDX, sBias + 128, sH, LDH, (float*)nullptr, 0, wid, lane);
    for (int j = tid; j < 1024; j += 256) {          // att2 weights (64x64)
        const int row = j >> 4, c4 = (j & 15) << 2;
        cvt_store_bf16x4(sW2 + row * LDH + c4, ((const float4*)a2w)[j]);
    }
    __syncthreads();

    // ---------------- Phase 4: s = relu(A @ att2^T + a2b) . att3_w ----------
    {
        const int half = lane >> 4;
        const int col  = lane & 15;
        for (int t = wid; t < 16; t += 8) {
            const int mt = t >> 2, nt = t & 3;
            v16bf av[2], bv[2];
#pragma unroll
            for (int k = 0; k < 2; ++k) {
                av[k] = lds_load_a(sH  + mt * 16 * LDH + k * 32, LDH, lane);
                bv[k] = lds_load_b(sW2 + nt * 16 * LDH + k * 32, LDH, lane);
            }
            v8f acc = {};
#pragma unroll
            for (int k = 0; k < 2; ++k)
                acc = __builtin_amdgcn_wmma_f32_16x16x32_bf16(
                    false, av[k], false, bv[k], (short)0, acc, false, false);

            const int   gn  = nt * 16 + col;
            const float bb  = sBias[192 + gn];
            const float a3v = sA3[gn];
#pragma unroll
            for (int r = 0; r < 8; ++r) {
                float v = acc[r] + bb;
                v = v > 0.f ? v : 0.f;
                const int gm = mt * 16 + r + 8 * half;
                atomicAdd(&sS[gm], v * a3v);         // ds_add_f32
            }
        }
    }
    __syncthreads();

    // ---------------- Phase 5: softmax over L=50 (att3_b cancels) -----------
    if (tid == 0) {
        float mx = -3.402823e38f;
        for (int l = 0; l < L_HIST; ++l) mx = sS[l] > mx ? sS[l] : mx;
        float sum = 0.f;
        for (int l = 0; l < L_HIST; ++l) {
            const float e = __expf(sS[l] - mx);
            sWt[l] = e;
            sum += e;
        }
        const float inv = 1.f / sum;
        for (int l = 0; l < L_HIST; ++l) sWt[l] *= inv;
    }
    __syncthreads();

    // ---------------- Phase 6: out[b,:] = sum_l w[l] * O[l,:] ---------------
    if (tid < DM) {
        float acc = 0.f;
        for (int l = 0; l < L_HIST; ++l) acc += sWt[l] * sOf[l * DM + tid];
        out[(size_t)b * DM + tid] = acc;
    }
}

extern "C" void kernel_launch(void* const* d_in, const int* in_sizes, int n_in,
                              void* d_out, int out_size, void* d_ws, size_t ws_size,
                              hipStream_t stream) {
    const int*   nodes = (const int*)  d_in[0];
    const int*   h_uv  = (const int*)  d_in[1];
    const int*   h_r   = (const int*)  d_in[2];
    // d_in[3] = history_uvt (unused by the reference)
    const float* v2e   = (const float*)d_in[4];
    const float* u2e   = (const float*)d_in[5];
    const float* r2e   = (const float*)d_in[6];
    const float* w1    = (const float*)d_in[7];
    const float* b1    = (const float*)d_in[8];
    const float* w2    = (const float*)d_in[9];
    const float* b2    = (const float*)d_in[10];
    const float* a1w   = (const float*)d_in[11];
    const float* a1b   = (const float*)d_in[12];
    const float* a2w   = (const float*)d_in[13];
    const float* a2b   = (const float*)d_in[14];
    const float* a3w   = (const float*)d_in[15];
    // d_in[16] = att3_b (constant shift; cancels in softmax)

    const int B = in_sizes[0];
    float* out = (float*)d_out;

    uv_aggregator_kernel<<<dim3(B), dim3(256), 0, stream>>>(
        nodes, h_uv, h_r, v2e, u2e, r2e,
        w1, b1, w2, b2, a1w, a1b, a2w, a2b, a3w, out);
}
